// FlatLogSoftmax_3702261809383
// MI455X (gfx1250) — matrix operand order repaired and checked by reading, not compiled, not run on hardware
//
#include <hip/hip_runtime.h>

typedef __bf16 bf16;
typedef __attribute__((ext_vector_type(4)))  __bf16 v4bf;
typedef __attribute__((ext_vector_type(8)))  __bf16 v8bf;
typedef __attribute__((ext_vector_type(16))) __bf16 v16bf;
typedef __attribute__((ext_vector_type(8)))  float  v8f;

#define BATCH   64
#define NNODES  512
#define NLEAVES 4096

// ---------------------------------------------------------------------------
// Kernel A: per-batch-row softmax stats. One 256-thread block per batch row.
// Produces P[b,l] = exp(x - rowmax) in bf16 (workspace) and logZ[b].
// ---------------------------------------------------------------------------
__global__ __launch_bounds__(256) void softmax_stats_kernel(
    const float* __restrict__ scores, void* __restrict__ Pv,
    float* __restrict__ logZ) {
  __shared__ float red[256];
  bf16* P = (bf16*)Pv;
  const int b = blockIdx.x;
  const int t = threadIdx.x;
  const float* row = scores + b * NLEAVES;

  float x[16];
  float m = -3.402823466e38f;
#pragma unroll
  for (int i = 0; i < 16; ++i) {
    x[i] = row[t + i * 256];
    m = fmaxf(m, x[i]);
  }
  red[t] = m;
  __syncthreads();
  for (int s = 128; s > 0; s >>= 1) {
    if (t < s) red[t] = fmaxf(red[t], red[t + s]);
    __syncthreads();
  }
  const float rowmax = red[0];
  __syncthreads();

  float z = 0.f;
#pragma unroll
  for (int i = 0; i < 16; ++i) {
    const float e = __expf(x[i] - rowmax);
    z += e;
    P[b * NLEAVES + t + i * 256] = (bf16)e;  // bf16 keeps f32 exponent range
  }
  red[t] = z;
  __syncthreads();
  for (int s = 128; s > 0; s >>= 1) {
    if (t < s) red[t] += red[t + s];
    __syncthreads();
  }
  if (t == 0) logZ[b] = __logf(red[0]);
}

// ---------------------------------------------------------------------------
// Kernel B: bool mask (1 byte/elem) -> bf16 0.0/1.0, 4 elems per thread.
// ---------------------------------------------------------------------------
__global__ __launch_bounds__(256) void mask_to_bf16_kernel(
    const unsigned char* __restrict__ mask, void* __restrict__ Mv) {
  v4bf* M = (v4bf*)Mv;
  const int i = blockIdx.x * 256 + threadIdx.x;
  const uchar4 m4 = ((const uchar4*)mask)[i];
  const bf16 one = (bf16)1.0f, zero = (bf16)0.0f;
  v4bf o;
  o[0] = m4.x ? one : zero;
  o[1] = m4.y ? one : zero;
  o[2] = m4.z ? one : zero;
  o[3] = m4.w ? one : zero;
  M[i] = o;
}

// ---------------------------------------------------------------------------
// Kernel C: split-K WMMA GEMM.  S[64x512] += P[64x4096] * mask_bf16^T.
//   blockIdx.x (0..31): kchunk = blk>>2 (8 chunks of K=512), mtile = blk&3
//                       -> K-loop bounds & trip count fully scalar/constant.
//   wave-in-block (0..7): ngroup -> 4 node tiles of 16 (A tile reused 4x).
// 2-stage software pipeline: iteration i+1's A/B tiles are loaded before
// iteration i's 4 WMMAs so global_load latency overlaps the matrix pipe.
// A layout (16x32 bf16): lane<16 -> row=lane, K 0-15; lane>=16 -> K 16-31.
// B layout (32x16 bf16): lane<16 -> col = mask row (contiguous), K 0-15;
//                        lane>=16 -> same col, K 16-31.
// ---------------------------------------------------------------------------
__device__ __forceinline__ v16bf cat8(v8bf x, v8bf y) {
  v16bf r;
#pragma unroll
  for (int i = 0; i < 8; ++i) { r[i] = x[i]; r[8 + i] = y[i]; }
  return r;
}

__global__ __launch_bounds__(256) void wmma_gemm_kernel(
    const void* __restrict__ Pv, const void* __restrict__ Mv,
    float* __restrict__ S) {
  const bf16* P = (const bf16*)Pv;
  const bf16* M = (const bf16*)Mv;

  const int kchunk = blockIdx.x >> 2;        // scalar: 8 K-chunks of 512
  const int mtile  = blockIdx.x & 3;         // scalar: 4 tiles of 16 rows
  const int ngroup = threadIdx.x >> 5;       // per-wave: 8 groups of 4 n-tiles
  const int lane = threadIdx.x & 31;
  const int hi = lane >> 4;
  const int lo = lane & 15;
  const int k0 = kchunk * 512;

  // Pre-offset base pointers (includes the per-half-wave K skew).
  const bf16* ap = P + (size_t)(mtile * 16 + lo) * NLEAVES + k0 + hi * 16;
  const bf16* bp0 = M + (size_t)(ngroup * 64 + 0 * 16 + lo) * NLEAVES + k0 + hi * 16;
  const bf16* bp1 = M + (size_t)(ngroup * 64 + 1 * 16 + lo) * NLEAVES + k0 + hi * 16;
  const bf16* bp2 = M + (size_t)(ngroup * 64 + 2 * 16 + lo) * NLEAVES + k0 + hi * 16;
  const bf16* bp3 = M + (size_t)(ngroup * 64 + 3 * 16 + lo) * NLEAVES + k0 + hi * 16;

#define LA(kk)    cat8(*(const v8bf*)(ap + (kk)), *(const v8bf*)(ap + (kk) + 8))
#define LB(p, kk) cat8(*(const v8bf*)((p) + (kk)), *(const v8bf*)((p) + (kk) + 8))

  v8f acc0 = {}, acc1 = {}, acc2 = {}, acc3 = {};

  // Stage 0 of the pipeline.
  v16bf a  = LA(0);
  v16bf b0 = LB(bp0, 0), b1 = LB(bp1, 0), b2 = LB(bp2, 0), b3 = LB(bp3, 0);

#pragma unroll
  for (int kk = 0; kk < 512; kk += 32) {
    v16bf an, b0n, b1n, b2n, b3n;
    const bool more = (kk + 32) < 512;       // compile-time after unroll
    if (more) {
      an  = LA(kk + 32);
      b0n = LB(bp0, kk + 32);
      b1n = LB(bp1, kk + 32);
      b2n = LB(bp2, kk + 32);
      b3n = LB(bp3, kk + 32);
    }
    acc0 = __builtin_amdgcn_wmma_f32_16x16x32_bf16(false, a, false, b0,
                                                   (short)0, acc0, false, false);
    acc1 = __builtin_amdgcn_wmma_f32_16x16x32_bf16(false, a, false, b1,
                                                   (short)0, acc1, false, false);
    acc2 = __builtin_amdgcn_wmma_f32_16x16x32_bf16(false, a, false, b2,
                                                   (short)0, acc2, false, false);
    acc3 = __builtin_amdgcn_wmma_f32_16x16x32_bf16(false, a, false, b3,
                                                   (short)0, acc3, false, false);
    if (more) { a = an; b0 = b0n; b1 = b1n; b2 = b2n; b3 = b3n; }
  }
#undef LA
#undef LB

  // C/D layout: VGPR r, lanes 0-15 -> M=r, lanes 16-31 -> M=r+8.
  v8f accs[4] = {acc0, acc1, acc2, acc3};
#pragma unroll
  for (int j = 0; j < 4; ++j) {
    const int col = ngroup * 64 + j * 16 + lo;
#pragma unroll
    for (int r = 0; r < 8; ++r) {
      const int orow = mtile * 16 + r + hi * 8;
      unsafeAtomicAdd(&S[orow * NNODES + col], accs[j][r]);
    }
  }
}

// ---------------------------------------------------------------------------
// Kernel D: out[b,n] = log(S[b,n]) - logZ[b]
// ---------------------------------------------------------------------------
__global__ __launch_bounds__(256) void epilogue_kernel(
    const float* __restrict__ S, const float* __restrict__ logZ,
    float* __restrict__ out) {
  const int i = blockIdx.x * 256 + threadIdx.x;  // 0..32767
  const int b = i >> 9;                          // /NNODES
  out[i] = __logf(S[i]) - logZ[b];
}

extern "C" void kernel_launch(void* const* d_in, const int* in_sizes, int n_in,
                              void* d_out, int out_size, void* d_ws, size_t ws_size,
                              hipStream_t stream) {
  const float*         scores = (const float*)d_in[0];
  const unsigned char* mask   = (const unsigned char*)d_in[1];  // jax bool: 1B/elem
  float*               out    = (float*)d_out;

  // Workspace layout (total ~4.65 MB):
  //   P    : 64*4096 bf16   = 512 KB
  //   Mb   : 512*4096 bf16  = 4 MB
  //   S    : 64*512 f32     = 128 KB
  //   logZ : 64 f32
  char*  ws   = (char*)d_ws;
  void*  P    = (void*)ws;
  void*  Mb   = (void*)(ws + (size_t)512 * 1024);
  float* S    = (float*)(ws + (size_t)512 * 1024 + (size_t)4 * 1024 * 1024);
  float* logZ = (float*)(ws + (size_t)512 * 1024 + (size_t)4 * 1024 * 1024 +
                         (size_t)128 * 1024);

  hipMemsetAsync(S, 0, (size_t)BATCH * NNODES * sizeof(float), stream);
  softmax_stats_kernel<<<BATCH, 256, 0, stream>>>(scores, P, logZ);
  mask_to_bf16_kernel<<<(NNODES * NLEAVES) / (256 * 4), 256, 0, stream>>>(mask, Mb);
  wmma_gemm_kernel<<<32, 256, 0, stream>>>(P, Mb, S);
  epilogue_kernel<<<(BATCH * NNODES) / 256, 256, 0, stream>>>(S, logZ, out);
}
